// TopNGenerator_56040733278762
// MI455X (gfx1250) — matrix operand order repaired and checked by reading, not compiled, not run on hardware
//
#include <hip/hip_runtime.h>

#define B_    256
#define LD_   256
#define HD_   512
#define CC_   64
#define SC_   256
#define PN_   100000
#define MAXN_ 128
#define CAT_  512          // SC+LD
#define M2_   (B_*MAXN_)   // 32768

typedef __bf16 v16bf __attribute__((ext_vector_type(16)));
typedef float  v8f   __attribute__((ext_vector_type(8)));
typedef int    v4i   __attribute__((vector_size(16)));
typedef __attribute__((address_space(1))) v4i* as1_v4i;
typedef __attribute__((address_space(3))) v4i* as3_v4i;

#if __has_builtin(__builtin_amdgcn_global_load_async_to_lds_b128)
#define HAVE_ASYNC 1
#endif

__device__ __forceinline__ unsigned short f2bf(float f) {
  unsigned u = __float_as_uint(f);
  unsigned r = u + 0x7FFFu + ((u >> 16) & 1u);   // round-to-nearest-even
  return (unsigned short)(r >> 16);
}
__device__ __forceinline__ unsigned okey(float f) {  // order-preserving uint key
  unsigned u = __float_as_uint(f);
  return (u & 0x80000000u) ? ~u : (u | 0x80000000u);
}

__device__ __forceinline__ void cp16(const unsigned short* g, unsigned short* l) {
#ifdef HAVE_ASYNC
  __builtin_amdgcn_global_load_async_to_lds_b128((as1_v4i)(g), (as3_v4i)(l), 0, 0);
#else
  *(uint4*)l = *(const uint4*)g;
#endif
}
__device__ __forceinline__ void cp_wait() {
#ifdef HAVE_ASYNC
#if __has_builtin(__builtin_amdgcn_s_wait_asynccnt)
  __builtin_amdgcn_s_wait_asynccnt(0);
#else
  asm volatile("s_wait_asynccnt 0x0" ::: "memory");
#endif
#endif
}

// ---------------- conversions ----------------
__global__ void k_cvt(const float* __restrict__ src, unsigned short* __restrict__ dst, int n) {
  int i = blockIdx.x * 256 + threadIdx.x;
  if (i < n) dst[i] = f2bf(src[i]);
}
// weight [K,N] f32 -> bf16 transposed [N,K]
__global__ void k_cvtT(const float* __restrict__ src, unsigned short* __restrict__ dst, int K, int N) {
  int i = blockIdx.x * 256 + threadIdx.x;
  if (i < K * N) {
    int n = i / K, k = i - n * K;
    dst[i] = f2bf(src[(size_t)k * N + n]);
  }
}

// ---------------- WMMA GEMM: C = relu?(A[M,K]bf16 @ Bt[N,K]bf16^T + bias) ----------------
// 128 threads (4 waves), block tile 128x64, BK=32, double-buffered async LDS staging.
// Wave w owns a 32x64 strip (rows w*16..w*16+16 and 64+w*16..64+w*16+16) -> 8 WMMA/K-step.
__global__ __launch_bounds__(128)
void k_gemm_bf16(const unsigned short* __restrict__ A,
                 const unsigned short* __restrict__ Bt,   // [N,K] bf16
                 const float* __restrict__ bias,
                 unsigned short* __restrict__ outB,
                 float* __restrict__ outF,
                 int M, int N, int K, int doRelu) {
  __shared__ __align__(16) unsigned short As[2][128 * 32];  // 8KB each
  __shared__ __align__(16) unsigned short Bs[2][64 * 32];   // 4KB each
  const int tid  = threadIdx.x;
  const int lane = tid & 31;
  const int wv   = tid >> 5;
  const int row0 = blockIdx.y * 128;
  const int col0 = blockIdx.x * 64;
  const int lrow = lane & 15;
  const int hi   = lane >> 4;    // 0 or 1

  // ---- hoisted staging descriptors (advance by k0 only) ----
  const unsigned short* aPtr[4]; int aOff[4];
#pragma unroll
  for (int j = 0; j < 4; ++j) {              // A: 128 rows x 32 k = 512 x 16B chunks
    int q = tid + j * 128, r = q >> 2, part = q & 3;
    aPtr[j] = A + (size_t)(row0 + r) * K + part * 8;
    aOff[j] = r * 32 + part * 8;
  }
  const unsigned short* bPtr[2]; int bOff[2]; bool bOk[2];
#pragma unroll
  for (int j = 0; j < 2; ++j) {              // B: 64 cols x 32 k = 256 x 16B chunks
    int q = tid + j * 128, r = q >> 2, part = q & 3;
    int col = col0 + r;
    bOk[j]  = col < N;
    bPtr[j] = Bt + (size_t)col * K + part * 8;
    bOff[j] = r * 32 + part * 8;
  }

  auto stage = [&](int buf, int k0) {
#pragma unroll
    for (int j = 0; j < 4; ++j) cp16(aPtr[j] + k0, &As[buf][aOff[j]]);
#pragma unroll
    for (int j = 0; j < 2; ++j) {
      if (bOk[j]) cp16(bPtr[j] + k0, &Bs[buf][bOff[j]]);
      else        *(uint4*)&Bs[buf][bOff[j]] = make_uint4(0u, 0u, 0u, 0u);
    }
  };

  v8f acc[2][4];
  const v8f vzero = {0.f,0.f,0.f,0.f,0.f,0.f,0.f,0.f};
#pragma unroll
  for (int s = 0; s < 2; ++s)
#pragma unroll
    for (int c = 0; c < 4; ++c) acc[s][c] = vzero;

  union Frag { uint4 q[2]; v16bf v; };

  stage(0, 0);
  cp_wait();
  __syncthreads();

  for (int k0 = 0; k0 < K; k0 += 32) {
    const int cur = (k0 >> 5) & 1;
    if (k0 + 32 < K) stage(cur ^ 1, k0 + 32);

    // load ALL fragments first, then issue 8 back-to-back WMMAs
    Frag fa[2];
#pragma unroll
    for (int s = 0; s < 2; ++s) {
      const unsigned short* ab = &As[cur][(s * 64 + wv * 16 + lrow) * 32];
      fa[s].q[0] = *(const uint4*)(ab + hi * 8);
      fa[s].q[1] = *(const uint4*)(ab + 16 + hi * 8);
    }
    Frag fb[4];
#pragma unroll
    for (int c = 0; c < 4; ++c) {
      const unsigned short* bb = &Bs[cur][(c * 16 + lrow) * 32];
      fb[c].q[0] = *(const uint4*)(bb + hi * 8);
      fb[c].q[1] = *(const uint4*)(bb + 16 + hi * 8);
    }
#pragma unroll
    for (int c = 0; c < 4; ++c) {
      acc[0][c] = __builtin_amdgcn_wmma_f32_16x16x32_bf16(
          false, fa[0].v, false, fb[c].v, (short)0, acc[0][c], false, false);
      acc[1][c] = __builtin_amdgcn_wmma_f32_16x16x32_bf16(
          false, fa[1].v, false, fb[c].v, (short)0, acc[1][c], false, false);
    }
    cp_wait();
    __syncthreads();
  }

#pragma unroll
  for (int s = 0; s < 2; ++s) {
#pragma unroll
    for (int c = 0; c < 4; ++c) {
#pragma unroll
      for (int r = 0; r < 8; ++r) {
        int row = row0 + s * 64 + wv * 16 + r + hi * 8;
        int col = col0 + c * 16 + lrow;
        if (col < N) {
          float v = acc[s][c][r] + (bias ? bias[col] : 0.0f);
          if (doRelu) v = fmaxf(v, 0.0f);
          size_t o = (size_t)row * N + col;
          if (outF) outF[o] = v;
          if (outB) outB[o] = f2bf(v);
        }
      }
    }
  }
}

// ---------------- L2-normalize ang rows, emit bf16 ----------------
__global__ void k_norm_ang(const float* __restrict__ ang, unsigned short* __restrict__ angB) {
  int row = blockIdx.x, tid = threadIdx.x;   // 64 threads
  __shared__ float red[64];
  float v = ang[row * CC_ + tid];
  red[tid] = v * v;
  __syncthreads();
  for (int s = 32; s > 0; s >>= 1) { if (tid < s) red[tid] += red[tid + s]; __syncthreads(); }
  float inv = 1.0f / (sqrtf(red[0]) + 1e-5f);
  angB[row * CC_ + tid] = f2bf(v * inv);
}

// ---------------- per-row softmax stats (max, sum-exp) ----------------
__global__ __launch_bounds__(256)
void k_rowstats(const float* __restrict__ logits, float* __restrict__ rowM, float* __restrict__ rowL) {
  int row = blockIdx.x, tid = threadIdx.x;
  __shared__ float red[256];
  const float* L = logits + (size_t)row * PN_;
  float m = -3.4e38f;
  for (int i = tid; i < PN_; i += 256) m = fmaxf(m, L[i]);
  red[tid] = m; __syncthreads();
  for (int s = 128; s > 0; s >>= 1) { if (tid < s) red[tid] = fmaxf(red[tid], red[tid + s]); __syncthreads(); }
  m = red[0]; __syncthreads();
  float sum = 0.f;
  for (int i = tid; i < PN_; i += 256) sum += expf(L[i] - m);
  red[tid] = sum; __syncthreads();
  for (int s = 128; s > 0; s >>= 1) { if (tid < s) red[tid] += red[tid + s]; __syncthreads(); }
  if (tid == 0) { rowM[row] = m; rowL[row] = red[0]; }
}

// ---------------- top-128 per row via 4-level radix select ----------------
__global__ __launch_bounds__(256)
void k_topk(const float* __restrict__ logits, const float* __restrict__ rowM,
            const float* __restrict__ rowL, float* __restrict__ tval, int* __restrict__ tidx) {
  const int row = blockIdx.x, tid = threadIdx.x;
  const float* L = logits + (size_t)row * PN_;
  __shared__ int hist[256];
  __shared__ unsigned sPrefix;
  __shared__ int sAbove;
  __shared__ float sval[MAXN_];
  __shared__ int   sidx[MAXN_];
  __shared__ int   eqbuf[MAXN_];
  __shared__ int   cntG, cntE;

  if (tid == 0) { sPrefix = 0u; sAbove = 0; }
  __syncthreads();
  for (int lvl = 0; lvl < 4; ++lvl) {
    hist[tid] = 0;
    __syncthreads();
    const int hb = 24 - 8 * lvl;
    const unsigned pfx = sPrefix;
    for (int i = tid; i < PN_; i += 256) {
      unsigned key = okey(L[i]);
      bool match = (lvl == 0) || ((key >> (hb + 8)) == pfx);
      if (match) atomicAdd(&hist[(key >> hb) & 0xFF], 1);
    }
    __syncthreads();
    if (tid == 0) {
      int cum = 0, chosen = 0, above = sAbove;
      for (int bin = 255; bin >= 0; --bin) {
        if (above + cum + hist[bin] >= MAXN_) { chosen = bin; break; }
        cum += hist[bin];
      }
      sAbove = above + cum;
      sPrefix = (pfx << 8) | (unsigned)chosen;
    }
    __syncthreads();
  }
  const unsigned T = sPrefix;
  if (tid == 0) { cntG = 0; cntE = 0; }
  __syncthreads();
  for (int i = tid; i < PN_; i += 256) {
    float v = L[i];
    unsigned key = okey(v);
    if (key > T) {
      int p = atomicAdd(&cntG, 1);
      if (p < MAXN_) { sval[p] = v; sidx[p] = i; }
    } else if (key == T) {
      int p = atomicAdd(&cntE, 1);
      if (p < MAXN_) eqbuf[p] = i;
    }
  }
  __syncthreads();
  int nG = min(cntG, MAXN_);
  int nE = min(cntE, MAXN_);
  int need = MAXN_ - nG;
  // sort equal-key candidate indices ascending (jax top_k takes lowest index first)
  for (int it = 0; it < MAXN_; ++it) {
    int i = (it & 1) + 2 * tid;
    if (tid < 64 && i + 1 < nE) {
      if (eqbuf[i] > eqbuf[i + 1]) { int t = eqbuf[i]; eqbuf[i] = eqbuf[i + 1]; eqbuf[i + 1] = t; }
    }
    __syncthreads();
  }
  if (tid < need) {
    if (tid < nE) { int id = eqbuf[tid]; sval[nG + tid] = L[id]; sidx[nG + tid] = id; }
    else          { sval[nG + tid] = -3.4e38f; sidx[nG + tid] = 0; }
  }
  __syncthreads();
  // sort 128 entries: value desc, index asc on tie (odd-even transposition)
  for (int it = 0; it < MAXN_; ++it) {
    int i = (it & 1) + 2 * tid;
    if (tid < 64 && i + 1 < MAXN_) {
      float a = sval[i], b = sval[i + 1];
      int ia = sidx[i], ib = sidx[i + 1];
      if (a < b || (a == b && ia > ib)) {
        sval[i] = b; sval[i + 1] = a; sidx[i] = ib; sidx[i + 1] = ia;
      }
    }
    __syncthreads();
  }
  if (tid < MAXN_) {
    float m = rowM[row], l = rowL[row];
    tval[(size_t)row * MAXN_ + tid] = expf(sval[tid] - m) / l;   // softmax prob
    tidx[(size_t)row * MAXN_ + tid] = sidx[tid];
  }
}

// ---------------- build cat = [alpha*sel+beta , latent] as bf16 ----------------
__global__ __launch_bounds__(128)
void k_build_cat(const float* __restrict__ tval, const int* __restrict__ tidx,
                 const float* __restrict__ points, const float* __restrict__ latent,
                 const float* __restrict__ l1w, const float* __restrict__ l1b,
                 const float* __restrict__ l2w, const float* __restrict__ l2b,
                 unsigned short* __restrict__ catB) {
  const int rowi = blockIdx.x;           // b*128 + j
  const int b = rowi >> 7;
  const float prob = tval[rowi];
  const int id = tidx[rowi];
  const float s = (float)MAXN_ * prob;
  for (int c = threadIdx.x; c < CAT_; c += 128) {
    float v;
    if (c < SC_) {
      float alpha = s * l1w[c] + l1b[c];
      float beta  = s * l2w[c] + l2b[c];
      v = alpha * points[(size_t)id * SC_ + c] + beta;
    } else {
      v = latent[(size_t)b * LD_ + (c - SC_)];
    }
    catB[(size_t)rowi * CAT_ + c] = f2bf(v);
  }
}

// ---------------- mask + n_r ----------------
__global__ void k_mask(const int* __restrict__ n, float* __restrict__ maskOut, float* __restrict__ nrOut) {
  int i = blockIdx.x * 256 + threadIdx.x;
  if (i < B_ * MAXN_) {
    int b = i >> 7, j = i & (MAXN_ - 1);
    int nr = min(max(n[b], 1), MAXN_);
    maskOut[i] = (j < nr) ? 1.0f : 0.0f;
  }
  if (i < B_) nrOut[i] = (float)min(max(n[i], 1), MAXN_);
}

// ---------------- host orchestration ----------------
extern "C" void kernel_launch(void* const* d_in, const int* in_sizes, int n_in,
                              void* d_out, int out_size, void* d_ws, size_t ws_size,
                              hipStream_t stream) {
  const float* latent  = (const float*)d_in[0];
  const int*   nArr    = (const int*)d_in[1];
  const float* points  = (const float*)d_in[2];
  const float* angles  = (const float*)d_in[3];
  const float* aw1 = (const float*)d_in[4];  const float* ab1 = (const float*)d_in[5];
  const float* aw2 = (const float*)d_in[6];  const float* ab2 = (const float*)d_in[7];
  const float* aw3 = (const float*)d_in[8];  const float* ab3 = (const float*)d_in[9];
  const float* aw4 = (const float*)d_in[10]; const float* ab4 = (const float*)d_in[11];
  const float* l1w = (const float*)d_in[12]; const float* l1b = (const float*)d_in[13];
  const float* l2w = (const float*)d_in[14]; const float* l2b = (const float*)d_in[15];
  const float* ow1 = (const float*)d_in[16]; const float* ob1 = (const float*)d_in[17];
  const float* ow2 = (const float*)d_in[18]; const float* ob2 = (const float*)d_in[19];
  const float* ow3 = (const float*)d_in[20]; const float* ob3 = (const float*)d_in[21];

  char* w = (char*)d_ws;
  // Region 0 (reused): logits f32 [256,100000] -> later cat/o1/o2 bf16
  float*          logits = (float*)w;
  unsigned short* catB   = (unsigned short*)w;                         // 32768*512*2 = 32MB
  unsigned short* o1B    = (unsigned short*)(w + 33554432);            // 32MB
  unsigned short* o2B    = (unsigned short*)(w + 67108864);            // 16MB
  size_t off = 102400000;                                              // logits bytes (256-aligned)
  auto alloc = [&](size_t bytes) -> char* {
    char* p = w + off; off = (off + bytes + 255) & ~(size_t)255; return p;
  };
  // all weights stored transposed [N,K] bf16 (WMMA-friendly; one-time cost)
  unsigned short* latB    = (unsigned short*)alloc((size_t)B_ * LD_ * 2);
  unsigned short* aw1B    = (unsigned short*)alloc((size_t)LD_ * HD_ * 2);
  unsigned short* aw2B    = (unsigned short*)alloc((size_t)HD_ * HD_ * 2);
  unsigned short* aw3B    = (unsigned short*)alloc((size_t)HD_ * HD_ * 2);
  unsigned short* aw4B    = (unsigned short*)alloc((size_t)HD_ * CC_ * 2);
  unsigned short* ow1B    = (unsigned short*)alloc((size_t)CAT_ * CAT_ * 2);
  unsigned short* ow2B    = (unsigned short*)alloc((size_t)CAT_ * SC_ * 2);
  unsigned short* ow3B    = (unsigned short*)alloc((size_t)SC_ * SC_ * 2);
  unsigned short* angTB   = (unsigned short*)alloc((size_t)CC_ * PN_ * 2);  // already [N,K]!
  unsigned short* h1B     = (unsigned short*)alloc((size_t)B_ * HD_ * 2);
  unsigned short* h2B     = (unsigned short*)alloc((size_t)B_ * HD_ * 2);
  unsigned short* h3B     = (unsigned short*)alloc((size_t)B_ * HD_ * 2);
  float*          angF    = (float*)alloc((size_t)B_ * CC_ * 4);
  unsigned short* angB    = (unsigned short*)alloc((size_t)B_ * CC_ * 2);
  float*          rowM    = (float*)alloc(B_ * 4);
  float*          rowL    = (float*)alloc(B_ * 4);
  float*          tval    = (float*)alloc((size_t)M2_ * 4);
  int*            tidx    = (int*)alloc((size_t)M2_ * 4);

  float* out    = (float*)d_out;                 // [32768, 256]
  float* maskO  = out + (size_t)M2_ * SC_;       // [32768]
  float* nrO    = maskO + M2_;                   // [256]

  auto cvtT = [&](const float* s, unsigned short* d, int K, int N) {
    k_cvtT<<<(K * N + 255) / 256, 256, 0, stream>>>(s, d, K, N);
  };
  k_cvt<<<(B_ * LD_ + 255) / 256, 256, 0, stream>>>(latent, latB, B_ * LD_);
  cvtT(aw1, aw1B, LD_, HD_);   cvtT(aw2, aw2B, HD_, HD_);
  cvtT(aw3, aw3B, HD_, HD_);   cvtT(aw4, aw4B, HD_, CC_);
  cvtT(ow1, ow1B, CAT_, CAT_); cvtT(ow2, ow2B, CAT_, SC_);
  cvtT(ow3, ow3B, SC_, SC_);
  k_cvt<<<(CC_ * PN_ + 255) / 256, 256, 0, stream>>>(angles, angTB, CC_ * PN_);

  // angle_mlp (M=256 -> grid.y = 2 with 128-row tiles)
  k_gemm_bf16<<<dim3(HD_/64, B_/128), 128, 0, stream>>>(latB, aw1B, ab1, h1B, nullptr, B_, HD_, LD_, 1);
  k_gemm_bf16<<<dim3(HD_/64, B_/128), 128, 0, stream>>>(h1B,  aw2B, ab2, h2B, nullptr, B_, HD_, HD_, 1);
  k_gemm_bf16<<<dim3(HD_/64, B_/128), 128, 0, stream>>>(h2B,  aw3B, ab3, h3B, nullptr, B_, HD_, HD_, 1);
  k_gemm_bf16<<<dim3(CC_/64, B_/128), 128, 0, stream>>>(h3B,  aw4B, ab4, nullptr, angF, B_, CC_, HD_, 0);
  k_norm_ang<<<B_, CC_, 0, stream>>>(angF, angB);

  // cosine logits [256, 100000]
  k_gemm_bf16<<<dim3((PN_ + 63) / 64, B_/128), 128, 0, stream>>>(
      angB, angTB, nullptr, nullptr, logits, B_, PN_, CC_, 0);

  k_rowstats<<<B_, 256, 0, stream>>>(logits, rowM, rowL);
  k_topk<<<B_, 256, 0, stream>>>(logits, rowM, rowL, tval, tidx);

  // cat (reuses logits region — top-k results are already extracted)
  k_build_cat<<<M2_, 128, 0, stream>>>(tval, tidx, points, latent, l1w, l1b, l2w, l2b, catB);

  // out_lin: the 30-GFLOP WMMA workload (grid.y = 256 with 128-row tiles)
  k_gemm_bf16<<<dim3(CAT_/64, M2_/128), 128, 0, stream>>>(catB, ow1B, ob1, o1B, nullptr, M2_, CAT_, CAT_, 1);
  k_gemm_bf16<<<dim3(SC_/64,  M2_/128), 128, 0, stream>>>(o1B,  ow2B, ob2, o2B, nullptr, M2_, SC_, CAT_, 1);
  k_gemm_bf16<<<dim3(SC_/64,  M2_/128), 128, 0, stream>>>(o2B,  ow3B, ob3, nullptr, out, M2_, SC_, SC_, 0);

  k_mask<<<(B_ * MAXN_ + 255) / 256, 256, 0, stream>>>(nArr, maskO, nrO);
  (void)in_sizes; (void)n_in; (void)out_size; (void)ws_size;
}
